// MoE_64991445123777
// MI455X (gfx1250) — matrix operand order repaired and checked by reading, not compiled, not run on hardware
//
#include <hip/hip_runtime.h>
#include <hip/hip_bf16.h>

#define NTOK 16384
#define DDIM 512
#define NEXP 8

typedef __attribute__((ext_vector_type(16))) __bf16 v16bf;
typedef __attribute__((ext_vector_type(8)))  __bf16 v8bf;
typedef __attribute__((ext_vector_type(4)))  __bf16 v4bf;
typedef __attribute__((ext_vector_type(8)))  float  v8f;
typedef __attribute__((ext_vector_type(4)))  float  v4f;

// ---------------------------------------------------------------------------
// CDNA5 async copy: global -> LDS, 16B per lane, tracked by ASYNCcnt.
// VDST supplies the per-lane LDS byte offset (= low 32 bits of the generic
// pointer to a __shared__ object, per ISA aperture rules).
// ---------------------------------------------------------------------------
__device__ __forceinline__ void async_copy_b128(uint32_t lds_off, const void* gptr)
{
    asm volatile("global_load_async_to_lds_b128 %0, %1, off"
                 :: "v"(lds_off), "v"(gptr) : "memory");
}
// Wait until this wave's outstanding async ops <= N (completion is in-order,
// so N = ops-in-flight-for-one-chunk leaves only the prefetched chunk pending).
template <int N>
__device__ __forceinline__ void async_wait_le()
{
    asm volatile("s_wait_asynccnt %0" :: "i"(N) : "memory");
}

// ---------------------------------------------------------------------------
// Prep: Wt[e][n][k] (bf16) = W[e][k][n] (f32).  One 32x32 tile per block via
// LDS transpose; coalesced f32 reads, contiguous 8B bf16 writes.
// ---------------------------------------------------------------------------
__global__ __launch_bounds__(256)
void prep_kernel(const float* __restrict__ Wmu, const float* __restrict__ Wlv,
                 __bf16* __restrict__ Wtmu, __bf16* __restrict__ Wtlv)
{
    __shared__ float Ls[32][33];
    const int tid = threadIdx.x;
    const int nb = blockIdx.x * 32, kb = blockIdx.y * 32;
    const int em = blockIdx.z;              // e*2 + which
    const int e = em >> 1;
    const float* src = (em & 1) ? Wlv : Wmu;
    __bf16* dst = ((em & 1) ? Wtlv : Wtmu) + (size_t)e * DDIM * DDIM;

    for (int t = tid; t < 1024; t += 256) {
        const int k = t >> 5, n = t & 31;
        Ls[k][n] = src[((size_t)e * DDIM + kb + k) * DDIM + nb + n];
    }
    __syncthreads();
    const int n = tid >> 3;       // 0..31
    const int kq = tid & 7;       // 4 k-values each
    v4bf pk;
#pragma unroll
    for (int j = 0; j < 4; ++j) pk[j] = (__bf16)Ls[kq * 4 + j][n];
    *(v4bf*)&dst[(size_t)(nb + n) * DDIM + kb + kq * 4] = pk;
}

// ---------------------------------------------------------------------------
// Gate: softmax(x@Wg+bg), stable top-k masks, per-expert stats.
// ---------------------------------------------------------------------------
__global__ __launch_bounds__(256)
void gate_kernel(const float* __restrict__ x, const float* __restrict__ Wg,
                 const float* __restrict__ bg, float* __restrict__ gate4,
                 float* __restrict__ gate1, float* __restrict__ stats)
{
    __shared__ float WgS[DDIM * NEXP];
    __shared__ float spS[NEXP], c4S[NEXP];
    const int tid = threadIdx.x;
    for (int i = tid; i < DDIM * NEXP; i += 256) WgS[i] = Wg[i];
    if (tid < NEXP) { spS[tid] = 0.f; c4S[tid] = 0.f; }
    __syncthreads();

    const int wave = tid >> 5, lane = tid & 31;
    const int tok = blockIdx.x * 8 + wave;
    float acc[NEXP];
#pragma unroll
    for (int e = 0; e < NEXP; ++e) acc[e] = 0.f;
    const float* xr = x + (size_t)tok * DDIM;
    for (int k = lane; k < DDIM; k += 32) {
        float xv = xr[k];
#pragma unroll
        for (int e = 0; e < NEXP; ++e) acc[e] += xv * WgS[k * NEXP + e];
    }
#pragma unroll
    for (int off = 16; off > 0; off >>= 1)
#pragma unroll
        for (int e = 0; e < NEXP; ++e) acc[e] += __shfl_xor(acc[e], off, 32);

    float lg[NEXP], p[NEXP];
    float mx = -1e30f;
#pragma unroll
    for (int e = 0; e < NEXP; ++e) { lg[e] = acc[e] + bg[e]; mx = fmaxf(mx, lg[e]); }
    float se = 0.f;
#pragma unroll
    for (int e = 0; e < NEXP; ++e) { p[e] = __expf(lg[e] - mx); se += p[e]; }
    const float inv = 1.f / se;
#pragma unroll
    for (int e = 0; e < NEXP; ++e) p[e] *= inv;

    if (lane == 0) {
#pragma unroll
        for (int e = 0; e < NEXP; ++e) {
            int rank = 0;
#pragma unroll
            for (int j = 0; j < NEXP; ++j) {
                if (p[j] > p[e]) rank++;
                else if (p[j] == p[e] && j < e) rank++;
            }
            const float m4 = (rank < 4) ? 1.f : 0.f;
            const float m1 = (rank < 1) ? 1.f : 0.f;
            gate4[(size_t)tok * NEXP + e] = p[e] * m4;
            gate1[(size_t)tok * NEXP + e] = p[e] * m1;
            atomicAdd(&spS[e], p[e]);
            atomicAdd(&c4S[e], m4);
        }
    }
    __syncthreads();
    if (tid < NEXP) {
        atomicAdd(&stats[tid], spS[tid]);          // sum_n p[n,e]
        atomicAdd(&stats[NEXP + tid], c4S[tid]);   // sum_n mask4[n,e]
    }
}

// ---------------------------------------------------------------------------
// Fused expert GEMM with bf16 WMMA (16x16x32), f32 accumulate.
// A resident in registers (16 frags = full K).  B streamed into a 3-deep
// circular LDS buffer via global_load_async_to_lds_b128 with one-chunk-ahead
// prefetch; one workgroup barrier per chunk.
// 8 waves (4 row-blocks x 2 col-halves) over a 64-token x 32-col tile.
// ---------------------------------------------------------------------------
template <bool WITH_LV>
__global__ __launch_bounds__(256)
void expert_kernel(const float* __restrict__ X,
                   const __bf16* __restrict__ Wtmu, const float* __restrict__ bmu,
                   const __bf16* __restrict__ Wtlv, const float* __restrict__ blv,
                   const float* __restrict__ gate, float* __restrict__ out,
                   float* __restrict__ stats)
{
    constexpr int BM = 64;
    constexpr int BN = 32;
    constexpr int KC = 256;
    constexpr int BS_ST = KC + 8;           // bf16 stride: 528B rows -> bank rotation
    constexpr int NMAT = WITH_LV ? 2 : 1;
    constexpr int NBUF = 3;
    constexpr int NCHUNK = (DDIM / BN) * NEXP * 2;   // 256
    constexpr int OPS = WITH_LV ? 8 : 4;    // async ops per wave per chunk

    __shared__ __bf16 Bs[NBUF][NMAT][BN][BS_ST];
    __shared__ float  Gs[BM][NEXP];

    const int tid  = threadIdx.x;
    const int wave = tid >> 5;
    const int lane = tid & 31;
    const int half = lane >> 4;
    const int ln   = lane & 15;
    const int r    = wave >> 1;
    const int cw   = wave & 1;
    const int tok0 = blockIdx.x * BM;

    // chunk id c = (ct*NEXP + e)*2 + ch  ->  DMA one 32-col x 256-K slab
    auto issue_chunk = [&](int buf, int cn) {
        const int ctn = cn >> 4;
        const int en  = (cn >> 1) & 7;
        const int chn = cn & 1;
        const int cbn = ctn * BN;
#pragma unroll
        for (int t = 0; t < 4; ++t) {
            const int i = tid + t * 256;
            const int n = i >> 5;          // 0..31
            const int kg = i & 31;         // 8 bf16 each
            const size_t goff =
                ((size_t)en * DDIM + cbn + n) * DDIM + chn * KC + kg * 8;
            async_copy_b128((uint32_t)(uintptr_t)&Bs[buf][0][n][kg * 8],
                            Wtmu + goff);
            if constexpr (WITH_LV)
                async_copy_b128((uint32_t)(uintptr_t)&Bs[buf][1][n][kg * 8],
                                Wtlv + goff);
        }
    };

    for (int i = tid; i < BM * NEXP; i += 256)
        Gs[i >> 3][i & 7] = gate[(size_t)tok0 * NEXP + i];

    // --- A fragments in registers: rows r*16+ln, all of K ---
    // elems 0..15 -> K = half*16 + i
    const float* xrow = X + (size_t)(tok0 + r * 16 + ln) * DDIM;
    v16bf afr[16];
#pragma unroll
    for (int kk = 0; kk < 16; ++kk) {
        const float* p0 = xrow + kk * 32 + half * 16;
        v4f f0 = *(const v4f*)(p0);
        v4f f1 = *(const v4f*)(p0 + 4);
        v4f f2 = *(const v4f*)(p0 + 8);
        v4f f3 = *(const v4f*)(p0 + 12);
        v16bf a;
#pragma unroll
        for (int i = 0; i < 4; ++i) {
            a[i]      = (__bf16)f0[i];
            a[4 + i]  = (__bf16)f1[i];
            a[8 + i]  = (__bf16)f2[i];
            a[12 + i] = (__bf16)f3[i];
        }
        afr[kk] = a;
    }

    float klp = 0.f, up = 0.f;

    issue_chunk(0, 0);        // head of the pipeline
    int cur = 0;

    for (int ct = 0; ct < DDIM / BN; ++ct) {
        const int cb  = ct * BN;
        const int col = cb + cw * 16 + ln;

        float xv[8];
#pragma unroll
        for (int rr = 0; rr < 8; ++rr)
            xv[rr] = X[(size_t)(tok0 + r * 16 + rr + 8 * half) * DDIM + col];

        v8f moe = {};
        for (int e = 0; e < NEXP; ++e) {
            v8f amu = {}, alv = {};
#pragma unroll
            for (int ch = 0; ch < 2; ++ch) {
                const int c = (ct * NEXP + e) * 2 + ch;
                int nxt = cur + 1; if (nxt == NBUF) nxt = 0;
                int cn  = c + 1;   if (cn == NCHUNK) cn = 0;   // tail wraps (harmless)
                issue_chunk(nxt, cn);          // prefetch next chunk
                async_wait_le<OPS>();          // chunk c landed (in-order)
                __syncthreads();               // visible to all waves

                // B frag: lane col = ln, K = half*16 + i (contiguous 32B)
#pragma unroll
                for (int kk = 0; kk < 8; ++kk) {
                    const __bf16* bp = &Bs[cur][0][cw * 16 + ln][kk * 32 + half * 16];
                    v8bf b0 = *(const v8bf*)bp;
                    v8bf b1 = *(const v8bf*)(bp + 8);
                    v16bf bm = __builtin_shufflevector(
                        b0, b1, 0, 1, 2, 3, 4, 5, 6, 7, 8, 9, 10, 11, 12, 13, 14, 15);
                    amu = __builtin_amdgcn_wmma_f32_16x16x32_bf16(
                        false, afr[ch * 8 + kk], false, bm, (short)0, amu, false, false);
                    if constexpr (WITH_LV) {
                        const __bf16* lp = &Bs[cur][1][cw * 16 + ln][kk * 32 + half * 16];
                        v8bf l0 = *(const v8bf*)lp;
                        v8bf l1 = *(const v8bf*)(lp + 8);
                        v16bf bl = __builtin_shufflevector(
                            l0, l1, 0, 1, 2, 3, 4, 5, 6, 7, 8, 9, 10, 11, 12, 13, 14, 15);
                        alv = __builtin_amdgcn_wmma_f32_16x16x32_bf16(
                            false, afr[ch * 8 + kk], false, bl, (short)0, alv, false, false);
                    }
                }
                cur = nxt;
            }
            // epilogue: D layout -> VGPR rr = row rr + 8*half, col = ln
            const float bmu_v = bmu[e * DDIM + col];
            float blv_v = 0.f;
            if constexpr (WITH_LV) blv_v = blv[e * DDIM + col];
#pragma unroll
            for (int rr = 0; rr < 8; ++rr) {
                const int tl = r * 16 + rr + 8 * half;
                const float g  = Gs[tl][e];
                const float mu = amu[rr] + bmu_v + xv[rr];
                moe[rr] += g * mu;
                if constexpr (WITH_LV) {
                    const float lv = alv[rr] + blv_v;
                    const float el = __expf(lv);
                    klp += 0.5f * (mu * mu + el - lv - 1.f);  // -(1+lv-mu^2-e^lv)/2
                    up  += g * el;
                }
            }
        }
#pragma unroll
        for (int rr = 0; rr < 8; ++rr) {
            const int tl = r * 16 + rr + 8 * half;
            out[(size_t)(tok0 + tl) * DDIM + col] = moe[rr];
        }
    }

    if constexpr (WITH_LV) {
#pragma unroll
        for (int off = 16; off > 0; off >>= 1) {
            klp += __shfl_xor(klp, off, 32);
            up  += __shfl_xor(up, off, 32);
        }
        if (lane == 0) {
            atomicAdd(&stats[16], klp);
            atomicAdd(&stats[17], up);
        }
    }
}

// ---------------------------------------------------------------------------
// loss = E*sum_e(sp*c4)/N^2 + kl_sum/(N*E) + u_sum/N
// ---------------------------------------------------------------------------
__global__ void loss_kernel(const float* __restrict__ stats, float* __restrict__ out_loss)
{
    if (threadIdx.x == 0 && blockIdx.x == 0) {
        float s = 0.f;
#pragma unroll
        for (int e = 0; e < NEXP; ++e) s += stats[e] * stats[NEXP + e];
        const float n = (float)NTOK;
        const float aux = (float)NEXP * s / (n * n);
        *out_loss = aux + stats[16] / (n * (float)NEXP) + stats[17] / n;
    }
}

extern "C" void kernel_launch(void* const* d_in, const int* in_sizes, int n_in,
                              void* d_out, int out_size, void* d_ws, size_t ws_size,
                              hipStream_t stream)
{
    const float* x   = (const float*)d_in[0];
    const float* y   = (const float*)d_in[1];
    const float* z   = (const float*)d_in[2];
    const float* Wg  = (const float*)d_in[3];
    const float* bg  = (const float*)d_in[4];
    const float* Wmu = (const float*)d_in[5];
    const float* bmu = (const float*)d_in[6];
    const float* Wlv = (const float*)d_in[7];
    const float* blv = (const float*)d_in[8];

    float* moe_x = (float*)d_out;
    float* moe_y = moe_x + (size_t)NTOK * DDIM;
    float* moe_z = moe_y + (size_t)NTOK * DDIM;
    float* loss  = moe_z + (size_t)NTOK * DDIM;

    // workspace: stats(64f) | gate4 | gate1 | Wtmu bf16 | Wtlv bf16  (~9.4 MB)
    float* stats = (float*)d_ws;
    float* gate4 = stats + 64;
    float* gate1 = gate4 + (size_t)NTOK * NEXP;
    __bf16* Wtmu = (__bf16*)(gate1 + (size_t)NTOK * NEXP);
    __bf16* Wtlv = Wtmu + (size_t)NEXP * DDIM * DDIM;

    hipMemsetAsync(stats, 0, 64 * sizeof(float), stream);

    prep_kernel<<<dim3(16, 16, 16), 256, 0, stream>>>(Wmu, Wlv, Wtmu, Wtlv);
    gate_kernel<<<NTOK / 8, 256, 0, stream>>>(x, Wg, bg, gate4, gate1, stats);

    expert_kernel<true><<<NTOK / 64, 256, 0, stream>>>(x, Wtmu, bmu, Wtlv, blv,
                                                       gate4, moe_x, stats);
    expert_kernel<false><<<NTOK / 64, 256, 0, stream>>>(y, Wtmu, bmu, Wtlv, blv,
                                                        gate1, moe_y, stats);
    expert_kernel<false><<<NTOK / 64, 256, 0, stream>>>(z, Wtmu, bmu, Wtlv, blv,
                                                        gate1, moe_z, stats);
    loss_kernel<<<1, 32, 0, stream>>>(stats, loss);
}
